// ScaledDotProductAttention_34815004902076
// MI455X (gfx1250) — matrix operand order repaired and checked by reading, not compiled, not run on hardware
//
#include <hip/hip_runtime.h>
#include <hip/hip_bf16.h>
#include <stdint.h>

// ---------------------------------------------------------------------------
// Flash-attention forward, B=16, L=4096, d=64, fp32 in/out, gfx1250.
//   - v_wmma_f32_16x16x32_f16 for QK^T and PV (f32 accumulate)
//   - K tile (contiguous 16KB) DMA'd by the Tensor Data Mover
//     (tensor_load_to_lds, TENSORcnt)
//   - f32->f16 conversion once per workgroup; WMMA fragments are pure
//     ds_load_b128 pairs
//   - softmax cross-lane reductions: fused v_permlane16_b32 + max/add
//     2-op asm butterflies (no tied-operand movs / canonicalizes)
// ---------------------------------------------------------------------------

typedef _Float16 v16h __attribute__((ext_vector_type(16)));
typedef _Float16 h8   __attribute__((ext_vector_type(8)));
typedef _Float16 h4   __attribute__((ext_vector_type(4)));
typedef float    v8f  __attribute__((ext_vector_type(8)));
typedef float    f4   __attribute__((ext_vector_type(4)));
typedef unsigned int u32x4 __attribute__((ext_vector_type(4)));
typedef int      i32x8 __attribute__((ext_vector_type(8)));
typedef int      i32x4 __attribute__((ext_vector_type(4)));

#define LSEQ 4096
#define DH   64
#define BQ   128              // 8 waves x 16 rows
#define BK   64               // key tile
#define NT   (LSEQ / BK)      // 64 tiles
#define VPADH 80              // padded f16 row for transposed V (32B-aligned rows)
#define QSCALE 0.18033688011112042f   // (1/sqrt(64)) * log2(e)

// 16-lane xor butterfly selectors for v_permlane16_b32
#define XOR1_LO 0x67452301u
#define XOR1_HI 0xEFCDAB89u
#define XOR2_LO 0x54761032u
#define XOR2_HI 0xDCFE98BAu
#define XOR4_LO 0x32107654u
#define XOR4_HI 0xBA98FEDCu
#define XOR8_LO 0xFEDCBA98u
#define XOR8_HI 0x76543210u

// one butterfly step: m = max(m, permlane16(m)) -- exactly 2 VALU ops
__device__ __forceinline__ float bfly_max(float m, unsigned s0, unsigned s1) {
  float t;
  asm("v_permlane16_b32 %0, %1, %2, %3\n\t"
      "v_max_num_f32 %1, %1, %0"
      : "=&v"(t), "+v"(m)
      : "s"(s0), "s"(s1));
  return m;
}
__device__ __forceinline__ float bfly_add(float s, unsigned s0, unsigned s1) {
  float t;
  asm("v_permlane16_b32 %0, %1, %2, %3\n\t"
      "v_add_f32 %1, %1, %0"
      : "=&v"(t), "+v"(s)
      : "s"(s0), "s"(s1));
  return s;
}
__device__ __forceinline__ float rowmax16(float m) {
  m = bfly_max(m, XOR1_LO, XOR1_HI);
  m = bfly_max(m, XOR2_LO, XOR2_HI);
  m = bfly_max(m, XOR4_LO, XOR4_HI);
  m = bfly_max(m, XOR8_LO, XOR8_HI);
  return m;
}
__device__ __forceinline__ float rowsum16(float s) {
  s = bfly_add(s, XOR1_LO, XOR1_HI);
  s = bfly_add(s, XOR2_LO, XOR2_HI);
  s = bfly_add(s, XOR4_LO, XOR4_HI);
  s = bfly_add(s, XOR8_LO, XOR8_HI);
  return s;
}

__global__ __launch_bounds__(256)
void fa_fwd_f16wmma(const float* __restrict__ qg,
                    const float* __restrict__ kg,
                    const float* __restrict__ vg,
                    float* __restrict__ og) {
  __shared__ float    sKraw[2][BK * DH];    // raw f32 K tile (TDM dest)
  __shared__ _Float16 sKh[BK * DH];         // converted f16 K tile [key][d]
  __shared__ _Float16 sVth[2][DH * VPADH];  // f16 transposed V [d][key]
  __shared__ _Float16 sP[8][16 * BK];       // per-wave P staging [m][key]

  const int tid  = threadIdx.x;
  const int lane = tid & 31;
  const int wid  = tid >> 5;
  const int hh   = lane >> 4;
  const int ln   = lane & 15;

  const int bidx  = blockIdx.x;
  const int batch = bidx >> 5;           // 32 q-tiles per batch
  const int qtile = bidx & 31;
  const int qrow0 = qtile * BQ + wid * 16;

  const size_t bbase = (size_t)batch * LSEQ * DH;

  // ---- Q fragments (A layout, 16x32 f16 x2), prescaled ----
  v16h aq[2];
  {
    const float* qrow = qg + bbase + (size_t)(qrow0 + ln) * DH;
#pragma unroll
    for (int t = 0; t < 2; ++t) {
      const int kb0 = t * 32 + hh * 8;
      const f4* q0 = (const f4*)(qrow + kb0);
      const f4* q1 = (const f4*)(qrow + kb0 + 16);
      f4 a0 = q0[0], a1 = q0[1], b0 = q1[0], b1 = q1[1];
#pragma unroll
      for (int e = 0; e < 4; ++e) {
        aq[t][e]      = (_Float16)(a0[e] * QSCALE);
        aq[t][4 + e]  = (_Float16)(a1[e] * QSCALE);
        aq[t][8 + e]  = (_Float16)(b0[e] * QSCALE);
        aq[t][12 + e] = (_Float16)(b1[e] * QSCALE);
      }
    }
  }

  // ---- per-tile load: K via TDM, V via f16 transpose ----
  auto issue_tile = [&](int tix, int buf) {
    // V: 4x4 (key x d) block per thread; convert to f16, packed b64 stores
    {
      const int kq = (tid & 15) * 4;
      const int dq = (tid >> 4) * 4;
      const float* vbase = vg + bbase + (size_t)tix * BK * DH;
      f4 r0 = *(const f4*)(vbase + (kq + 0) * DH + dq);
      f4 r1 = *(const f4*)(vbase + (kq + 1) * DH + dq);
      f4 r2 = *(const f4*)(vbase + (kq + 2) * DH + dq);
      f4 r3 = *(const f4*)(vbase + (kq + 3) * DH + dq);
#pragma unroll
      for (int e = 0; e < 4; ++e) {
        h4 pk = { (_Float16)r0[e], (_Float16)r1[e],
                  (_Float16)r2[e], (_Float16)r3[e] };
        *(h4*)&sVth[buf][(dq + e) * VPADH + kq] = pk;
      }
    }
    // K: contiguous 16KB tile -> LDS via Tensor Data Mover (wave 0 only)
    {
      uint32_t ldsb = (uint32_t)(uintptr_t)&sKraw[buf][0];
      const float* gsrc = kg + bbase + (size_t)tix * BK * DH;
#if __has_builtin(__builtin_amdgcn_tensor_load_to_lds)
      if (wid == 0) {
        // D# group0: count=1 | lds_addr | 57b global addr | type=2
        uint64_t ga = (uint64_t)(uintptr_t)gsrc;
        u32x4 g0 = { 1u, ldsb, (unsigned)(ga & 0xffffffffu),
                     (unsigned)(ga >> 32) | 0x80000000u };
        // D# group1: data_size=4B; 1D tile: tensor_dim0=tile_dim0=4096
        i32x8 g1 = { 0x20000, 0x10000000, 0x00010000, 0x10000000,
                     0, 4096, 0, 0 };
        i32x4 gz4 = { 0, 0, 0, 0 };
        i32x8 gz8 = { 0, 0, 0, 0, 0, 0, 0, 0 };
        __builtin_amdgcn_tensor_load_to_lds(g0, g1, gz4, gz4, gz8, 0);
      }
#else
#pragma unroll
      for (int j = 0; j < 4; ++j) {
        uint32_t off = (uint32_t)(tid * 16 + j * 4096);
        asm volatile("global_load_async_to_lds_b128 %0, %1, %2"
                     :: "v"(ldsb + off), "v"(off), "s"((const char*)gsrc)
                     : "memory");
      }
#endif
    }
  };

  v8f oacc[4];
  float rowm[8], rowl[8];
#pragma unroll
  for (int n = 0; n < 4; ++n) oacc[n] = (v8f){0,0,0,0,0,0,0,0};
#pragma unroll
  for (int vv = 0; vv < 8; ++vv) { rowm[vv] = -__builtin_inff(); rowl[vv] = 0.f; }

  issue_tile(0, 0);

#pragma unroll 1
  for (int it = 0; it < NT; ++it) {
    const int cur = it & 1;

#if __has_builtin(__builtin_amdgcn_tensor_load_to_lds)
    if (wid == 0) __builtin_amdgcn_s_wait_tensorcnt(0);
#elif __has_builtin(__builtin_amdgcn_s_wait_asynccnt)
    __builtin_amdgcn_s_wait_asynccnt(0);
#else
    asm volatile("s_wait_asynccnt 0" ::: "memory");
#endif
    __syncthreads();   // raw K(it) + Vt(it) resident; compute(it-1) retired

    if (it + 1 < NT) issue_tile(it + 1, cur ^ 1);

    // ---- cooperative K f32 -> f16 convert (once per workgroup) ----
    {
      const f4* src = (const f4*)&sKraw[cur][tid * 16];
      f4 x0 = src[0], x1 = src[1], x2 = src[2], x3 = src[3];
      h8 lo, hi;
#pragma unroll
      for (int e = 0; e < 4; ++e) {
        lo[e]     = (_Float16)x0[e];
        lo[4 + e] = (_Float16)x1[e];
        hi[e]     = (_Float16)x2[e];
        hi[4 + e] = (_Float16)x3[e];
      }
      *(h8*)&sKh[tid * 16]     = lo;
      *(h8*)&sKh[tid * 16 + 8] = hi;
    }
    __syncthreads();   // sKh ready

    // ---- S = Q K^T : four 16x16 fragments (pure ds loads, no cvt) ----
    v8f sfr[4];
#pragma unroll
    for (int c = 0; c < 4; ++c) {
      v8f acc = (v8f){0,0,0,0,0,0,0,0};
#pragma unroll
      for (int t = 0; t < 2; ++t) {
        h8 blo = *(const h8*)&sKh[(c * 16 + ln) * DH + t * 32 + hh * 8];
        h8 bhi = *(const h8*)&sKh[(c * 16 + ln) * DH + t * 32 + hh * 8 + 16];
        v16h bf = __builtin_shufflevector(blo, bhi, 0,1,2,3,4,5,6,7,8,9,10,11,12,13,14,15);
        acc = __builtin_amdgcn_wmma_f32_16x16x32_f16(false, aq[t], false, bf,
                                                     (short)0, acc, false, false);
      }
      sfr[c] = acc;
    }

    // ---- online softmax (exp2 domain); row = vv + 8*hh ----
#pragma unroll
    for (int vv = 0; vv < 8; ++vv) {
      float m = fmaxf(fmaxf(sfr[0][vv], sfr[1][vv]), fmaxf(sfr[2][vv], sfr[3][vv]));
      m = rowmax16(m);
      float nm    = fmaxf(rowm[vv], m);
      float alpha = __builtin_amdgcn_exp2f(rowm[vv] - nm);
      rowm[vv] = nm;
      float ps = 0.f;
#pragma unroll
      for (int c = 0; c < 4; ++c) {
        float p = __builtin_amdgcn_exp2f(sfr[c][vv] - nm);
        sfr[c][vv] = p;
        ps += p;
      }
      ps = rowsum16(ps);
      rowl[vv] = rowl[vv] * alpha + ps;
#pragma unroll
      for (int c = 0; c < 4; ++c) oacc[c][vv] *= alpha;
#pragma unroll
      for (int c = 0; c < 4; ++c)
        sP[wid][(vv + 8 * hh) * BK + c * 16 + ln] = (_Float16)sfr[c][vv];
    }

    // ---- reload P as A fragments (wave-private; DS in-order) ----
    v16h pa[2];
#pragma unroll
    for (int t = 0; t < 2; ++t) {
      const int kb0 = t * 32 + hh * 8;
      h8 lo = *(const h8*)&sP[wid][ln * BK + kb0];
      h8 hi = *(const h8*)&sP[wid][ln * BK + kb0 + 16];
#pragma unroll
      for (int e = 0; e < 8; ++e) { pa[t][e] = lo[e]; pa[t][8 + e] = hi[e]; }
    }

    // ---- O += P V : B fragments contiguous in transposed f16 V ----
#pragma unroll
    for (int n = 0; n < 4; ++n) {
#pragma unroll
      for (int t = 0; t < 2; ++t) {
        h8 blo = *(const h8*)&sVth[cur][(n * 16 + ln) * VPADH + t * 32 + hh * 8];
        h8 bhi = *(const h8*)&sVth[cur][(n * 16 + ln) * VPADH + t * 32 + hh * 8 + 16];
        v16h bf = __builtin_shufflevector(blo, bhi, 0,1,2,3,4,5,6,7,8,9,10,11,12,13,14,15);
        oacc[n] = __builtin_amdgcn_wmma_f32_16x16x32_f16(false, pa[t], false, bf,
                                                         (short)0, oacc[n], false, false);
      }
    }
  }

  // ---- epilogue: normalize and store ----
#pragma unroll
  for (int vv = 0; vv < 8; ++vv) {
    float inv = 1.0f / rowl[vv];
    float* orow = og + bbase + (size_t)(qrow0 + vv + 8 * hh) * DH;
#pragma unroll
    for (int n = 0; n < 4; ++n)
      orow[n * 16 + ln] = oacc[n][vv] * inv;
  }
}

extern "C" void kernel_launch(void* const* d_in, const int* in_sizes, int n_in,
                              void* d_out, int out_size, void* d_ws, size_t ws_size,
                              hipStream_t stream) {
  (void)in_sizes; (void)n_in; (void)out_size; (void)d_ws; (void)ws_size;
  const float* q = (const float*)d_in[0];
  const float* k = (const float*)d_in[1];
  const float* v = (const float*)d_in[2];
  float* out = (float*)d_out;
  dim3 grid(16 * (LSEQ / BQ));   // 512 workgroups
  fa_fwd_f16wmma<<<grid, 256, 0, stream>>>(q, k, v, out);
}
